// Net_27891517620299
// MI455X (gfx1250) — matrix operand order repaired and checked by reading, not compiled, not run on hardware
//
#include <hip/hip_runtime.h>
#include <hip/hip_bf16.h>
#include <stdint.h>

typedef __attribute__((ext_vector_type(16))) _Float16 v16h;
typedef __attribute__((ext_vector_type(8)))  _Float16 v8h;
typedef __attribute__((ext_vector_type(2)))  _Float16 v2h;
typedef __attribute__((ext_vector_type(8)))  float    v8f;

#define DEV __device__ __forceinline__

// ---------------- persistent per-call stats block in workspace ----------------
struct Stats {
  unsigned gmax_u, gmin_u;                 // pooled-input min/max (flipped)
  float q0_scale, q0_mn, q0_zpmin;         // input quant params
  unsigned thr_u[4];                       // per-layer |score| median (uint-ordered)
  float hwsum[4];                          // sum(hw_one) per layer
  float inv_s3, inv_s4;                    // 1/scale for fc layers (conv scales cancel in BN)
  float ch1_sum[64], ch1_sumsq[64];
  unsigned ch1_max[64], ch1_min[64];
  float bn1_mean[64], bn1_invsd[64];
  float q1_scale;
  float ch2_sum[16], ch2_sumsq[16];
  unsigned ch2_max[16], ch2_min[16];
  float bn2_mean[16], bn2_invsd[16];
  float q2_scale;
  unsigned fc1_max_u, fc1_min_u;
  float q3_scale;
};

DEV unsigned flipf(float f){ unsigned u=__float_as_uint(f); return (u&0x80000000u)? ~u : (u|0x80000000u); }
DEV float unflipf(unsigned u){ return __uint_as_float((u&0x80000000u)? (u&0x7FFFFFFFu) : ~u); }

// async copy 16B: per-lane global addr -> per-lane LDS addr, tracked by ASYNCcnt
DEV void async_b128(const void* gptr, void* lptr){
  uint64_t ga = (uint64_t)(size_t)gptr;
  unsigned la = (unsigned)(size_t)lptr;     // low 32 bits of generic LDS ptr = LDS offset
  asm volatile("global_load_async_to_lds_b128 %0, %1, off" :: "v"(la), "v"(ga) : "memory");
}
DEV void async_b128_off16(const void* gptr, void* lptr){
  uint64_t ga = (uint64_t)(size_t)gptr;
  unsigned la = (unsigned)(size_t)lptr;
  asm volatile("global_load_async_to_lds_b128 %0, %1, off offset:16" :: "v"(la), "v"(ga) : "memory");
}
DEV void wait_async0(){ asm volatile("s_wait_asynccnt 0" ::: "memory"); }

// ---------------- WMMA fragment builders (ISA 7.12.2, 16-bit layouts) ----------------
DEV v16h frag_A(const _Float16* base, int ldk){
  int l  = threadIdx.x & 31;
  int m  = l & 15;
  int k0 = (l & 16) ? 8 : 0;
  const _Float16* p = base + m*ldk + k0;
  v8h lo = *(const v8h*)(p);
  v8h hi = *(const v8h*)(p + 16);
  v16h r;
#pragma unroll
  for (int i=0;i<8;i++){ r[i]=lo[i]; r[8+i]=hi[i]; }
  return r;
}
DEV v16h frag_B(const _Float16* base, int ldk){
  int l  = threadIdx.x & 31;
  int n  = l & 15;
  int k0 = (l & 16) ? 8 : 0;
  const _Float16* p = base + n*ldk + k0;
  v8h lo = *(const v8h*)(p);
  v8h hi = *(const v8h*)(p + 16);
  v16h r;
#pragma unroll
  for (int i=0;i<8;i++){ r[i]=lo[i]; r[8+i]=hi[i]; }
  return r;
}
DEV v8f wmma16(v16h a, v16h b, v8f c){
  return __builtin_amdgcn_wmma_f32_16x16x32_f16(false, a, false, b, (short)0, c, false, false);
}

// ---------------- init ----------------
__global__ void init_kernel(Stats* st, _Float16* w1q){
  unsigned* p = (unsigned*)st;
  int nw = (int)(sizeof(Stats)/4);
  for (int i=threadIdx.x; i<nw; i+=1024) p[i]=0u;
  unsigned* q = (unsigned*)w1q;                 // 64*32 halfs (zero = K padding)
  for (int i=threadIdx.x; i<1024; i+=1024) q[i]=0u;
  __syncthreads();
  int t = threadIdx.x;
  if (t < 64)       st->ch1_min[t]    = 0xFFFFFFFFu;
  else if (t < 80)  st->ch2_min[t-64] = 0xFFFFFFFFu;
  else if (t == 80) st->gmin_u        = 0xFFFFFFFFu;
  else if (t == 81) st->fc1_min_u     = 0xFFFFFFFFu;
}

// ---------------- stage 0: maxpool 28->14 + global min/max ----------------
__global__ void pool_minmax_kernel(const float* __restrict__ x, float* __restrict__ pooled, Stats* st){
  __shared__ unsigned s_mx, s_mn;
  int tid = threadIdx.x;
  if (tid==0){ s_mx=0u; s_mn=0xFFFFFFFFu; }
  __syncthreads();
  int i = blockIdx.x*256 + tid;                 // < 4096*196 exactly
  int b = i/196, p = i%196, py = p/14, px = p%14;
  const float* ip = x + (size_t)b*784 + (2*py)*28 + 2*px;
  float v = fmaxf(fmaxf(ip[0],ip[1]), fmaxf(ip[28],ip[29]));
  pooled[i] = v;
  atomicMax(&s_mx, flipf(v));
  atomicMin(&s_mn, flipf(v));
  __syncthreads();
  if (tid==0){ atomicMax(&st->gmax_u, s_mx); atomicMin(&st->gmin_u, s_mn); }
}

// ---------------- stage 1: radix-select median of |scores| per layer ----------------
__global__ void radix_select_kernel(const float* s1, const float* s2, const float* s3, const float* s4, Stats* st){
  const float* ptrs[4] = {s1,s2,s3,s4};
  const int    ns[4]   = {576, 9216, 51200, 1280};
  int layer = blockIdx.x;
  const float* sc = ptrs[layer];
  int n = ns[layer];
  __shared__ unsigned hist[256];
  __shared__ unsigned sh_prefix;
  __shared__ int sh_rank;
  if (threadIdx.x==0){ sh_prefix=0u; sh_rank = n/2; }
  __syncthreads();
  for (int shift=24; shift>=0; shift-=8){
    hist[threadIdx.x]=0u;
    __syncthreads();
    unsigned mask = (shift==24)?0u:(0xFFFFFFFFu << (shift+8));
    unsigned pref = sh_prefix;
    for (int i=threadIdx.x; i<n; i+=256){
      unsigned u = __float_as_uint(fabsf(sc[i]));
      if ((u & mask)==pref) atomicAdd(&hist[(u>>shift)&255], 1u);
    }
    __syncthreads();
    if (threadIdx.x==0){
      unsigned cum=0; int rank=sh_rank; unsigned b=0;
      for (; b<256; b++){ if (cum + hist[b] > (unsigned)rank) break; cum += hist[b]; }
      sh_prefix = pref | (b<<shift);
      sh_rank   = rank - (int)cum;
    }
    __syncthreads();
  }
  if (threadIdx.x==0) st->thr_u[layer] = sh_prefix;
}

// ---------------- stage 2: blend weights -> f16, accumulate sum(hw_one) ----------------
__global__ void blend_kernel(const float* __restrict__ sc, const float* __restrict__ sg,
                             const float* __restrict__ one_, const float* __restrict__ zero_,
                             _Float16* __restrict__ dst, int n, int mode, int layer, Stats* st){
  __shared__ float ssum;
  if (threadIdx.x==0) ssum = 0.f;
  __syncthreads();
  int i = blockIdx.x*256 + threadIdx.x;
  float one = 0.f;
  if (i < n){
    unsigned u = __float_as_uint(fabsf(sc[i]));
    float m = (u >= st->thr_u[layer]) ? 1.f : 0.f;
    one = one_[i];
    float w = sg[i]*(one*m + zero_[i]*(1.f-m));
    int di = (mode==1) ? ((i/9)*32 + (i%9)) : i;        // conv1: pad K 9->32
    dst[di] = (_Float16)w;
  }
  atomicAdd(&ssum, one);
  __syncthreads();
  if (threadIdx.x==0) atomicAdd(&st->hwsum[layer], ssum);
}

__global__ void finalize_early_kernel(Stats* st, const int* qb){
  if (threadIdx.x) return;
  float levels = (float)((1<<qb[0]) - 1);
  float mx = unflipf(st->gmax_u), mn = unflipf(st->gmin_u);
  float s  = (mx - mn)/levels;
  st->q0_scale = s; st->q0_mn = mn; st->q0_zpmin = floorf(mn/s);
  float std3 = sqrtf(2.f)/sqrtf(400.f*0.5f);
  float std4 = sqrtf(2.f)/sqrtf(128.f*0.5f);
  st->inv_s3 = std3 * 51200.f / st->hwsum[2];
  st->inv_s4 = std4 * 1280.f  / st->hwsum[3];
}

// ---------------- stage 3: quantize pooled input -> f16 ----------------
__global__ void quant_pool_kernel(const float* __restrict__ pooled, _Float16* __restrict__ qpooled,
                                  const Stats* st){
  int i = blockIdx.x*256 + threadIdx.x;
  float s = st->q0_scale;
  float v = (rintf((pooled[i]-st->q0_mn)/s) + st->q0_zpmin)*s;
  qpooled[i] = (_Float16)v;
}

// ---------------- conv1: WMMA implicit GEMM, M=589824 N=64 K=9(pad 32) + BN1 stats ----------------
__global__ void __launch_bounds__(256) conv1_kernel(const _Float16* __restrict__ qpooled,
                                                    const _Float16* __restrict__ w1q,
                                                    _Float16* __restrict__ act1, Stats* st){
  __shared__ alignas(16) _Float16 As[128*32];
  __shared__ alignas(16) _Float16 Bs[64*32];
  __shared__ float s_sum[64], s_sq[64];
  __shared__ unsigned s_mx[64], s_mn[64];
  int tid = threadIdx.x;
  async_b128(w1q + tid*8, Bs + tid*8);                  // 2048 halfs, K>=9 pre-zeroed
  if (tid < 64){ s_sum[tid]=0.f; s_sq[tid]=0.f; s_mx[tid]=0u; s_mn[tid]=0xFFFFFFFFu; }
  int m0 = blockIdx.x*128;
  {
    int m = tid>>1, kh = tid&1;
    int gm = m0 + m;
    int b = gm/144, pos = gm%144;
    int oy = pos/12, ox = pos%12;
    const _Float16* ip = qpooled + b*196;
#pragma unroll
    for (int j=0;j<16;j++){
      int k = kh*16 + j;
      _Float16 v = (_Float16)0.f;
      if (k < 9){ int ky=k/3, kx=k%3; v = ip[(oy+ky)*14 + (ox+kx)]; }
      As[m*32 + k] = v;
    }
  }
  wait_async0();
  __syncthreads();
  int wave = tid>>5, lane = tid&31;
  v16h a = frag_A(As + wave*16*32, 32);
  v8f acc[4];
#pragma unroll
  for (int nt=0; nt<4; nt++){
    v16h b = frag_B(Bs + nt*16*32, 32);
    v8f z = {};
    acc[nt] = wmma16(a, b, z);
  }
  int n_lo = lane & 15, mhi = (lane>>4)*8;
#pragma unroll
  for (int nt=0; nt<4; nt++){
    int c = nt*16 + n_lo;
    float lsum=0.f, lsq=0.f, lmx=-3.4e38f, lmn=3.4e38f;
#pragma unroll
    for (int r=0;r<8;r++){
      int m = wave*16 + mhi + r;
      int gm = m0 + m;
      int b = gm/144, pos = gm%144;
      float v = acc[nt][r];
      act1[(b*64 + c)*144 + pos] = (_Float16)v;
      lsum += v; lsq += v*v; lmx = fmaxf(lmx,v); lmn = fminf(lmn,v);
    }
    atomicAdd(&s_sum[c], lsum);
    atomicAdd(&s_sq[c],  lsq);
    atomicMax(&s_mx[c],  flipf(lmx));
    atomicMin(&s_mn[c],  flipf(lmn));
  }
  __syncthreads();
  if (tid < 64){
    atomicAdd(&st->ch1_sum[tid],   s_sum[tid]);
    atomicAdd(&st->ch1_sumsq[tid], s_sq[tid]);
    atomicMax(&st->ch1_max[tid],   s_mx[tid]);
    atomicMin(&st->ch1_min[tid],   s_mn[tid]);
  }
}

// ---------------- finalize BN params + post-ReLU quant scale ----------------
__global__ void finalize_bn_kernel(Stats* st, const int* qb, int layer){
  __shared__ float smx[64], smn[64];
  int C = (layer==1)?64:16;
  float Ninv = (layer==1)? (1.f/(4096.f*144.f)) : (1.f/(4096.f*25.f));
  float* sum   = (layer==1)? st->ch1_sum   : st->ch2_sum;
  float* sumsq = (layer==1)? st->ch1_sumsq : st->ch2_sumsq;
  unsigned* mxu= (layer==1)? st->ch1_max   : st->ch2_max;
  unsigned* mnu= (layer==1)? st->ch1_min   : st->ch2_min;
  float* mean  = (layer==1)? st->bn1_mean  : st->bn2_mean;
  float* invsd = (layer==1)? st->bn1_invsd : st->bn2_invsd;
  int c = threadIdx.x;
  if (c < C){
    float m  = sum[c]*Ninv;
    float v  = sumsq[c]*Ninv - m*m;
    float is = rsqrtf(v + 1e-5f);
    mean[c]=m; invsd[c]=is;
    float bmx = (unflipf(mxu[c]) - m)*is; bmx = bmx>0.f?bmx:0.f;
    float bmn = (unflipf(mnu[c]) - m)*is; bmn = bmn>0.f?bmn:0.f;
    smx[c]=bmx; smn[c]=bmn;
  }
  __syncthreads();
  if (c==0){
    float MX=smx[0], MN=smn[0];
    for (int k=1;k<C;k++){ MX=fmaxf(MX,smx[k]); MN=fminf(MN,smn[k]); }
    float levels=(float)((1<<qb[0])-1);
    float s=(MX-MN)/levels;
    if (layer==1) st->q1_scale=s; else st->q2_scale=s;
  }
}

// ---------------- conv2: fused BN1+ReLU+quant on load; WMMA GEMM M=409600 N=16 K=576 ----------------
__global__ void __launch_bounds__(256) conv2_kernel(const _Float16* __restrict__ act1,
                                                    const _Float16* __restrict__ w2q,
                                                    _Float16* __restrict__ act2,
                                                    const Stats* st, const int* qb){
  __shared__ alignas(16) _Float16 As[128*32];
  __shared__ alignas(16) _Float16 Bs[16*32];
  __shared__ float s_mean[64], s_invsd[64];
  int tid = threadIdx.x;
  if (tid < 64){ s_mean[tid]=st->bn1_mean[tid]; s_invsd[tid]=st->bn1_invsd[tid]; }
  float qs1 = st->q1_scale;
  float inv_qs1 = 1.f/qs1;
  float levels = (float)((1<<qb[0])-1);
  int m0 = blockIdx.x*128;
  int m = tid>>1, kh = tid&1;
  int gm = m0 + m;
  int bimg = gm/100, pos = gm%100;
  int oy = pos/10, ox = pos%10;
  const _Float16* ibase = act1 + (size_t)bimg*64*144;
  v8f acc = {};
  __syncthreads();
  for (int kt=0; kt<18; kt++){
    if (tid < 64){
      int n = tid>>2, sub = (tid&3)*8;
      async_b128(w2q + n*576 + kt*32 + sub, Bs + n*32 + sub);
    }
    if (kt < 17) __builtin_prefetch(ibase + (((kt+1)*32)/9)*144, 0, 0);
#pragma unroll
    for (int j=0;j<16;j++){
      int k  = kt*32 + kh*16 + j;
      int ic = k/9, rem = k - ic*9, ky = rem/3, kx = rem%3;
      float xv = (float)ibase[ic*144 + (oy+ky)*12 + (ox+kx)];
      float bv = (xv - s_mean[ic])*s_invsd[ic];            // BN1
      bv = bv>0.f?bv:0.f;                                  // ReLU
      float q = rintf(bv*inv_qs1);                         // quant (after_relu)
      q = q<0.f?0.f:(q>levels?levels:q);
      As[m*32 + kh*16 + j] = (_Float16)(q*qs1);
    }
    wait_async0();
    __syncthreads();
    v16h a = frag_A(As + (tid>>5)*16*32, 32);
    v16h b = frag_B(Bs, 32);
    acc = wmma16(a, b, acc);
    __syncthreads();
  }
  int wave = tid>>5, lane = tid&31;
  int c = lane & 15, mhi = (lane>>4)*8;
#pragma unroll
  for (int r=0;r<8;r++){
    int mm = wave*16 + mhi + r;
    int g2 = m0 + mm;
    int bi = g2/100, ps = g2%100;
    act2[(bi*16 + c)*100 + ps] = (_Float16)acc[r];
  }
}

// ---------------- maxpool2 (10->5) + BN2 stats ----------------
__global__ void pool2_kernel(const _Float16* __restrict__ act2, float* __restrict__ pooled2, Stats* st){
  __shared__ float ssum[16], ssq[16];
  __shared__ unsigned smx[16], smn[16];
  int tid = threadIdx.x;
  if (tid < 16){ ssum[tid]=0.f; ssq[tid]=0.f; smx[tid]=0u; smn[tid]=0xFFFFFFFFu; }
  __syncthreads();
  int i = blockIdx.x*256 + tid;                         // < 1638400 exactly
  int b = i/400, r = i%400, c = r/25, p = r%25;
  int py = p/5, px = p%5;
  const _Float16* ip = act2 + (b*16 + c)*100 + (2*py)*10 + 2*px;
  float v = fmaxf(fmaxf((float)ip[0],(float)ip[1]), fmaxf((float)ip[10],(float)ip[11]));
  pooled2[i] = v;
  atomicAdd(&ssum[c], v);
  atomicAdd(&ssq[c],  v*v);
  atomicMax(&smx[c],  flipf(v));
  atomicMin(&smn[c],  flipf(v));
  __syncthreads();
  if (tid < 16){
    atomicAdd(&st->ch2_sum[tid],   ssum[tid]);
    atomicAdd(&st->ch2_sumsq[tid], ssq[tid]);
    atomicMax(&st->ch2_max[tid],   smx[tid]);
    atomicMin(&st->ch2_min[tid],   smn[tid]);
  }
}

// ---------------- BN2+ReLU+quant -> x3q (flatten layout matches reshape) ----------------
__global__ void bnq2_kernel(const float* __restrict__ pooled2, _Float16* __restrict__ x3q,
                            const Stats* st, const int* qb){
  int i = blockIdx.x*256 + threadIdx.x;                 // < 1638400 exactly
  int c = (i/25) & 15;
  float b = (pooled2[i] - st->bn2_mean[c])*st->bn2_invsd[c];
  b = b>0.f?b:0.f;
  float s = st->q2_scale;
  float levels = (float)((1<<qb[0])-1);
  float q = rintf(b/s); q = q<0.f?0.f:(q>levels?levels:q);
  x3q[i] = (_Float16)(q*s);
}

// ---------------- fc1: WMMA GEMM M=4096 N=128 K=400(pad 416), async tile loads ----------------
__global__ void __launch_bounds__(256) fc1_kernel(const _Float16* __restrict__ x3q,
                                                  const _Float16* __restrict__ w3q,
                                                  float* __restrict__ fc1raw, Stats* st){
  __shared__ alignas(16) _Float16 As[128*32];
  __shared__ alignas(16) _Float16 Bs[128*32];
  __shared__ unsigned s_mx, s_mn;
  int tid = threadIdx.x;
  if (tid==0){ s_mx=0u; s_mn=0xFFFFFFFFu; }
  int m0 = blockIdx.x*128;
  int row = tid>>1, kh = tid&1;
  v8f acc[8];
  v8f zero = {};
#pragma unroll
  for (int nt=0; nt<8; nt++) acc[nt]=zero;
  for (int kt=0; kt<13; kt++){
    bool full = (kt < 12) || (kh == 0);                 // kh==0 @ kt==12 covers k=384..399
    if (full){
      const _Float16* gA = x3q + (m0+row)*400 + kt*32 + kh*16;
      const _Float16* gB = w3q + row*400      + kt*32 + kh*16;
      _Float16* lA = As + row*32 + kh*16;
      _Float16* lB = Bs + row*32 + kh*16;
      async_b128(gA, lA);
      async_b128_off16(gA, lA);
      async_b128(gB, lB);
      async_b128_off16(gB, lB);
    } else {
      v8h z = {};
      *(v8h*)(As + row*32 + 16) = z; *(v8h*)(As + row*32 + 24) = z;
      *(v8h*)(Bs + row*32 + 16) = z; *(v8h*)(Bs + row*32 + 24) = z;
    }
    wait_async0();
    __syncthreads();
    v16h a = frag_A(As + (tid>>5)*16*32, 32);
#pragma unroll
    for (int nt=0; nt<8; nt++){
      v16h b = frag_B(Bs + nt*16*32, 32);
      acc[nt] = wmma16(a, b, acc[nt]);
    }
    __syncthreads();
  }
  float inv_s3 = st->inv_s3;
  int wave = tid>>5, lane = tid&31;
  int n_lo = lane & 15, mhi = (lane>>4)*8;
  float lmx=-3.4e38f, lmn=3.4e38f;
#pragma unroll
  for (int nt=0; nt<8; nt++){
    int n = nt*16 + n_lo;
#pragma unroll
    for (int r=0;r<8;r++){
      int m = wave*16 + mhi + r;
      float v = acc[nt][r]*inv_s3;
      v = v>0.f?v:0.f;
      fc1raw[(m0+m)*128 + n] = v;
      lmx = fmaxf(lmx,v); lmn = fminf(lmn,v);
    }
  }
  atomicMax(&s_mx, flipf(lmx));
  atomicMin(&s_mn, flipf(lmn));
  __syncthreads();
  if (tid==0){ atomicMax(&st->fc1_max_u, s_mx); atomicMin(&st->fc1_min_u, s_mn); }
}

__global__ void finalize3_kernel(Stats* st, const int* qb){
  if (threadIdx.x) return;
  float levels = (float)((1<<qb[0])-1);
  float mx = unflipf(st->fc1_max_u), mn = unflipf(st->fc1_min_u);
  st->q3_scale = (mx - mn)/levels;
}

// ---------------- fc2: quantize-on-load + WMMA GEMM M=4096 N=10(pad16) K=128 ----------------
__global__ void __launch_bounds__(256) fc2_kernel(const float* __restrict__ fc1raw,
                                                  const _Float16* __restrict__ w4q,
                                                  float* __restrict__ logits,
                                                  const Stats* st, const int* qb){
  __shared__ alignas(16) _Float16 As[128*32];
  __shared__ alignas(16) _Float16 Bs[16*32];
  int tid = threadIdx.x;
  int m0 = blockIdx.x*128;
  float qs3 = st->q3_scale, inv_s4 = st->inv_s4;
  float levels = (float)((1<<qb[0])-1);
  int row = tid>>1, kh = tid&1;
  if (tid < 192) Bs[320 + tid] = (_Float16)0.f;         // zero pad rows n=10..15 once
  v8f acc = {};
  for (int kt=0; kt<4; kt++){
    if (tid < 40){                                      // n=0..9, 4x16B per row
      int n = tid>>2, sub = (tid&3)*8;
      async_b128(w4q + n*128 + kt*32 + sub, Bs + n*32 + sub);
    }
#pragma unroll
    for (int j=0;j<16;j++){
      int k = kt*32 + kh*16 + j;
      float x = fc1raw[(m0+row)*128 + k];
      float q = rintf(x/qs3); q = q<0.f?0.f:(q>levels?levels:q);
      As[row*32 + kh*16 + j] = (_Float16)(q*qs3);
    }
    wait_async0();
    __syncthreads();
    v16h a = frag_A(As + (tid>>5)*16*32, 32);
    v16h b = frag_B(Bs, 32);
    acc = wmma16(a, b, acc);
    __syncthreads();
  }
  int wave = tid>>5, lane = tid&31;
  int n = lane & 15, mhi = (lane>>4)*8;
#pragma unroll
  for (int r=0;r<8;r++){
    int m = wave*16 + mhi + r;
    logits[(m0+m)*16 + n] = acc[r]*inv_s4;
  }
}

// ---------------- log_softmax over 10 classes ----------------
__global__ void logsoftmax_kernel(const float* __restrict__ logits, float* __restrict__ out){
  int r = blockIdx.x*256 + threadIdx.x;                 // < 4096 exactly
  const float* p = logits + r*16;
  float mx = p[0];
#pragma unroll
  for (int j=1;j<10;j++) mx = fmaxf(mx, p[j]);
  float s = 0.f;
#pragma unroll
  for (int j=0;j<10;j++) s += expf(p[j]-mx);
  float ls = logf(s);
#pragma unroll
  for (int j=0;j<10;j++) out[r*10 + j] = p[j] - mx - ls;
}

// ---------------- host-side launch ----------------
static inline size_t aup(size_t x){ return (x + 255) & ~(size_t)255; }

extern "C" void kernel_launch(void* const* d_in, const int* in_sizes, int n_in,
                              void* d_out, int out_size, void* d_ws, size_t ws_size,
                              hipStream_t stream) {
  (void)in_sizes; (void)n_in; (void)out_size; (void)ws_size;
  const float* x       = (const float*)d_in[0];
  const float* sc[4]   = {(const float*)d_in[1],(const float*)d_in[5],(const float*)d_in[9], (const float*)d_in[13]};
  const float* sg[4]   = {(const float*)d_in[2],(const float*)d_in[6],(const float*)d_in[10],(const float*)d_in[14]};
  const float* one[4]  = {(const float*)d_in[3],(const float*)d_in[7],(const float*)d_in[11],(const float*)d_in[15]};
  const float* zero[4] = {(const float*)d_in[4],(const float*)d_in[8],(const float*)d_in[12],(const float*)d_in[16]};
  const int*   qb      = (const int*)d_in[17];
  float* out = (float*)d_out;

  char* w = (char*)d_ws;
  size_t o = 0;
  Stats*    st      = (Stats*)(w+o);     o = aup(o + sizeof(Stats));
  float*    pooled  = (float*)(w+o);     o = aup(o + (size_t)4096*196*4);
  _Float16* qpooled = (_Float16*)(w+o);  o = aup(o + (size_t)4096*196*2);
  _Float16* act1    = (_Float16*)(w+o);  o = aup(o + (size_t)4096*64*144*2);
  _Float16* act2    = (_Float16*)(w+o);  o = aup(o + (size_t)4096*16*100*2);
  float*    pooled2 = (float*)(w+o);     o = aup(o + (size_t)4096*400*4);
  _Float16* x3q     = (_Float16*)(w+o);  o = aup(o + (size_t)4096*400*2);
  float*    fc1raw  = (float*)(w+o);     o = aup(o + (size_t)4096*128*4);
  float*    logits  = (float*)(w+o);     o = aup(o + (size_t)4096*16*4);
  _Float16* w1q     = (_Float16*)(w+o);  o = aup(o + (size_t)64*32*2);
  _Float16* w2q     = (_Float16*)(w+o);  o = aup(o + (size_t)16*576*2);
  _Float16* w3q     = (_Float16*)(w+o);  o = aup(o + (size_t)128*400*2);
  _Float16* w4q     = (_Float16*)(w+o);  o = aup(o + (size_t)10*128*2);

  init_kernel<<<1, 1024, 0, stream>>>(st, w1q);
  pool_minmax_kernel<<<3136, 256, 0, stream>>>(x, pooled, st);
  radix_select_kernel<<<4, 256, 0, stream>>>(sc[0], sc[1], sc[2], sc[3], st);
  blend_kernel<<<3,   256, 0, stream>>>(sc[0], sg[0], one[0], zero[0], w1q,   576, 1, 0, st);
  blend_kernel<<<36,  256, 0, stream>>>(sc[1], sg[1], one[1], zero[1], w2q,  9216, 0, 1, st);
  blend_kernel<<<200, 256, 0, stream>>>(sc[2], sg[2], one[2], zero[2], w3q, 51200, 0, 2, st);
  blend_kernel<<<5,   256, 0, stream>>>(sc[3], sg[3], one[3], zero[3], w4q,  1280, 0, 3, st);
  finalize_early_kernel<<<1, 32, 0, stream>>>(st, qb);
  quant_pool_kernel<<<3136, 256, 0, stream>>>(pooled, qpooled, st);
  conv1_kernel<<<4608, 256, 0, stream>>>(qpooled, w1q, act1, st);
  finalize_bn_kernel<<<1, 64, 0, stream>>>(st, qb, 1);
  conv2_kernel<<<3200, 256, 0, stream>>>(act1, w2q, act2, st, qb);
  pool2_kernel<<<6400, 256, 0, stream>>>(act2, pooled2, st);
  finalize_bn_kernel<<<1, 64, 0, stream>>>(st, qb, 2);
  bnq2_kernel<<<6400, 256, 0, stream>>>(pooled2, x3q, st, qb);
  fc1_kernel<<<32, 256, 0, stream>>>(x3q, w3q, fc1raw, st);
  finalize3_kernel<<<1, 32, 0, stream>>>(st, qb);
  fc2_kernel<<<32, 256, 0, stream>>>(fc1raw, w4q, logits, st, qb);
  logsoftmax_kernel<<<16, 256, 0, stream>>>(logits, out);
}
